// RoPEMultiheadAttention_21629455302947
// MI455X (gfx1250) — matrix operand order repaired and checked by reading, not compiled, not run on hardware
//
#include <hip/hip_runtime.h>
#include <hip/hip_bf16.h>

// ---------------------------------------------------------------------------
// RoPE multi-head attention for MI455X (gfx1250), wave32 + WMMA f16->f32.
//   B=4, S=2048, D=1024, H=16, Dh=64
// Pipeline: f32->f16 | QKV WMMA-GEMM (+bias, scatter) | RoPE | flash-attn
//           (WMMA QK^T + online softmax + WMMA PV, row-sums via P@ones WMMA)
//           | out-proj WMMA-GEMM.
// CDNA5 specifics: v_wmma_f32_16x16x32_f16, global_load_async_to_lds_b128
// staging (ASYNCcnt), explicit s_wait_dscnt for the in-wave P round trip.
// ---------------------------------------------------------------------------

typedef __attribute__((ext_vector_type(16))) _Float16 v16h;
typedef __attribute__((ext_vector_type(8)))  float    v8f;

#define BATCH   4
#define SEQ     2048
#define DMODEL  1024
#define NHEADS  16
#define HEADDIM 64

union HalfVec {
    v16h v;
    uint4 q[2];
    _Float16 h[16];
};

__device__ __forceinline__ v8f v8f_zero() {
    v8f z = {0.f, 0.f, 0.f, 0.f, 0.f, 0.f, 0.f, 0.f};
    return z;
}

// A-fragment (16x32 f16): lane%16 = M row; lane/16 selects K interleave group.
// Per ISA 7.12.2: half idx 0..7 -> K = g*8 + idx ; idx 8..15 -> K = 16+g*8+(idx-8).
__device__ __forceinline__ v16h load_frag_a(const _Float16* p, int g) {
    HalfVec f;
    f.q[0] = *reinterpret_cast<const uint4*>(p + g * 8);
    f.q[1] = *reinterpret_cast<const uint4*>(p + 16 + g * 8);
    return f.v;
}

// B-fragment (32x16 f16) from [k][n] LDS (stride 72): lane%16 = N column;
// idx 0..7 -> K = g*8 + idx ; idx 8..15 -> K = 16+g*8+(idx-8).
__device__ __forceinline__ v16h load_frag_b(const _Float16* p, int g) {
    HalfVec f;
#pragma unroll
    for (int i = 0; i < 8; ++i) {
        f.h[i]     = p[(g * 8 + i) * 72];
        f.h[8 + i] = p[(16 + g * 8 + i) * 72];
    }
    return f.v;
}

__device__ __forceinline__ v8f wmma_f16(v16h a, v16h b, v8f c) {
    return __builtin_amdgcn_wmma_f32_16x16x32_f16(
        /*neg_a=*/false, a, /*neg_b=*/false, b,
        /*c_mod=*/(short)0, c, /*reuse_a=*/false, /*reuse_b=*/false);
}

// Async global->LDS copies (CDNA5, ASYNCcnt).  The instruction's immediate
// offset is added to BOTH the global address and the LDS address:
__device__ __forceinline__ void async_copy32(unsigned ldsoff, unsigned goff,
                                             unsigned long long base) {
    asm volatile(
        "global_load_async_to_lds_b128 %0, %1, %2\n\t"
        "global_load_async_to_lds_b128 %0, %1, %2 offset:16"
        :
        : "v"(ldsoff), "v"(goff), "s"(base)
        : "memory");
}

__device__ __forceinline__ void async_copy64(unsigned ldsoff, unsigned goff,
                                             unsigned long long base) {
    asm volatile(
        "global_load_async_to_lds_b128 %0, %1, %2\n\t"
        "global_load_async_to_lds_b128 %0, %1, %2 offset:16\n\t"
        "global_load_async_to_lds_b128 %0, %1, %2 offset:32\n\t"
        "global_load_async_to_lds_b128 %0, %1, %2 offset:48"
        :
        : "v"(ldsoff), "v"(goff), "s"(base)
        : "memory");
}

__device__ __forceinline__ void wait_async0() {
#if __has_builtin(__builtin_amdgcn_s_wait_asynccnt)
    __builtin_amdgcn_s_wait_asynccnt(0);
#else
    asm volatile("s_wait_asynccnt 0x0" ::: "memory");
#endif
}

// Generic pointer to a __shared__ object: low 32 bits = wave-relative LDS
// byte address (LDS aperture maps addr[31:0] directly).
__device__ __forceinline__ unsigned lds_addr32(const void* p) {
    return (unsigned)(unsigned long long)p;
}

// ---------------------------------------------------------------------------
// fp32 -> fp16 convert
// ---------------------------------------------------------------------------
__global__ void cvt_f32_f16_kernel(const float* __restrict__ src,
                                   _Float16* __restrict__ dst, int n) {
    int i = blockIdx.x * blockDim.x + threadIdx.x;
    if (i < n) dst[i] = (_Float16)src[i];
}

// ---------------------------------------------------------------------------
// GEMM: C[M,N] = A[M,K] @ W[N,K]^T + bias[N]
//   MODE 0: store fp32 to outF [M,N]
//   MODE 1: store fp16 scattered into q/k/v  [B,H,S,64]  (N = 3*DMODEL)
// Block: 256 threads (8 waves). Block tile 128(M) x 64(N), K step 32.
// Wave w owns rows [m0+32w, m0+32w+32) (2 A-frags), all 64 N cols.
// ---------------------------------------------------------------------------
template <int MODE>
__global__ __launch_bounds__(256) void wmma_gemm_kernel(
    const _Float16* __restrict__ A, const _Float16* __restrict__ W,
    const float* __restrict__ bias, int M, int N, int K,
    float* __restrict__ outF,
    _Float16* __restrict__ qh, _Float16* __restrict__ kh,
    _Float16* __restrict__ vh) {
    __shared__ alignas(16) _Float16 Asm[128 * 40];  // [m][k] 128x32, stride 40
    __shared__ alignas(16) _Float16 Wtm[32 * 72];   // [k][n] 32x64, stride 72

    const int tid  = threadIdx.x;
    const int wave = tid >> 5;
    const int lane = tid & 31;
    const int ln   = lane & 15;
    const int g    = lane >> 4;
    const int n0   = blockIdx.x * 64;
    const int m0   = blockIdx.y * 128;

    v8f acc[2][4];
#pragma unroll
    for (int mt = 0; mt < 2; ++mt)
#pragma unroll
        for (int t = 0; t < 4; ++t) acc[mt][t] = v8f_zero();

    float biasv[4];
#pragma unroll
    for (int t = 0; t < 4; ++t) biasv[t] = bias[n0 + t * 16 + ln];

    // Per-thread staging coordinates (invariant over K loop).
    const int arow = tid >> 1, acb = (tid & 1) * 16;  // A: 2 thr/row, 32B each
    const int nrow = tid >> 2, wcb = (tid & 3) * 8;   // W: 4 thr/row, 16B each
    const unsigned aldsoff = lds_addr32(&Asm[arow * 40 + acb]);
    const unsigned long long abase = (unsigned long long)A;

    for (int k0 = 0; k0 < K; k0 += 32) {
        // A tile 128x32, natural layout: async global->LDS, no VGPR data.
        async_copy32(aldsoff,
                     (unsigned)(((size_t)(m0 + arow) * K + k0 + acb) * 2),
                     abase);
        // W tile 64(n)x32(k) transposed -> Wtm[k][n] (scatter, manual).
        {
            union { uint4 q; _Float16 h[8]; } tmp;
            tmp.q = *reinterpret_cast<const uint4*>(
                W + (size_t)(n0 + nrow) * K + k0 + wcb);
#pragma unroll
            for (int i = 0; i < 8; ++i) Wtm[(wcb + i) * 72 + nrow] = tmp.h[i];
        }
        wait_async0();
        __syncthreads();

        v16h a0 = load_frag_a(&Asm[(wave * 32 + ln) * 40], g);
        v16h a1 = load_frag_a(&Asm[(wave * 32 + 16 + ln) * 40], g);
#pragma unroll
        for (int t = 0; t < 4; ++t) {
            v16h b = load_frag_b(&Wtm[t * 16 + ln], g);
            acc[0][t] = wmma_f16(a0, b, acc[0][t]);
            acc[1][t] = wmma_f16(a1, b, acc[1][t]);
        }
        __syncthreads();
    }

    // Epilogue. C layout: VGPR r -> row r (lanes 0-15) / r+8 (lanes 16-31),
    // col = lane%16. All tensor/head/batch selection is block-uniform.
    if (MODE == 0) {
        float* ob = outF + (size_t)(m0 + wave * 32 + g * 8) * N + n0 + ln;
#pragma unroll
        for (int mt = 0; mt < 2; ++mt)
#pragma unroll
            for (int t = 0; t < 4; ++t)
#pragma unroll
                for (int r = 0; r < 8; ++r)
                    ob[(size_t)(mt * 16 + r) * N + t * 16] =
                        acc[mt][t][r] + biasv[t];
    } else {
        const int which = n0 >> 10;          // 0=q 1=k 2=v (tile inside one)
        const int hh    = (n0 & 1023) >> 6;  // head (tile inside one head)
        const int bb    = m0 >> 11;          // batch (tile inside one batch)
        const int s0    = (m0 & 2047) + wave * 32 + g * 8;
        _Float16* dst = ((which == 0) ? qh : (which == 1) ? kh : vh) +
                        (((size_t)bb * NHEADS + hh) * SEQ + s0) * HEADDIM +
                        ln;
#pragma unroll
        for (int mt = 0; mt < 2; ++mt)
#pragma unroll
            for (int t = 0; t < 4; ++t)
#pragma unroll
                for (int r = 0; r < 8; ++r)
                    dst[(size_t)(mt * 16 + r) * HEADDIM + t * 16] =
                        (_Float16)(acc[mt][t][r] + biasv[t]);
    }
}

// ---------------------------------------------------------------------------
// RoPE in-place on q and k, layout [B,H,S,64]; pairs (d, d+32), d<32.
//   inv_freq = 10000^(-d/32) = exp(-d * ln(10000)/32)
// ---------------------------------------------------------------------------
__global__ void rope_kernel(_Float16* __restrict__ qh,
                            _Float16* __restrict__ kh) {
    const size_t per = (size_t)BATCH * NHEADS * SEQ * 32;  // 4,194,304
    size_t idx = (size_t)blockIdx.x * blockDim.x + threadIdx.x;
    if (idx >= 2 * per) return;
    _Float16* ptr = (idx < per) ? qh : kh;
    size_t i = (idx < per) ? idx : idx - per;
    int d      = (int)(i & 31);
    size_t row = i >> 5;  // b*H*S + h*S + s
    int s      = (int)(row & (SEQ - 1));
    float inv_freq = __expf((float)d * -0.28782313662425572f);  // -ln(1e4)/32
    float ang = (float)s * inv_freq;
    float c, sn;
    __sincosf(ang, &sn, &c);
    _Float16* base = ptr + row * HEADDIM;
    float x1 = (float)base[d];
    float x2 = (float)base[d + 32];
    base[d]      = (_Float16)(x1 * c - x2 * sn);
    base[d + 32] = (_Float16)(x1 * sn + x2 * c);
}

// ---------------------------------------------------------------------------
// Flash attention. Grid: (S/64, B*H). Block: 128 threads (4 waves).
// Wave w: query rows [q0+16w, q0+16w+16). Key tile = 64 keys per step.
// K staged transposed in LDS ([dim][key]) for QK^T B-frags (manual scatter);
// V staged natural ([key][dim]) via async global->LDS for PV B-frags.
// Per 64-key step: 8 QK^T WMMAs, 2 row-sum WMMAs (P @ ones), 8 PV WMMAs,
// one 4-stage max butterfly (ds_bpermute) -- sum butterflies eliminated.
// ---------------------------------------------------------------------------
__global__ __launch_bounds__(128) void flash_attn_kernel(
    const _Float16* __restrict__ qh, const _Float16* __restrict__ kh,
    const _Float16* __restrict__ vh, _Float16* __restrict__ attn) {
    __shared__ alignas(16) _Float16 Kt[64 * 72];      // [dim(64)][key(64)+pad]
    __shared__ alignas(16) _Float16 Vs[64 * 72];      // [key(64)][dim(64)+pad]
    __shared__ alignas(16) _Float16 Ps[4 * 16 * 72];  // per-wave 16x64 P tile

    const int tid  = threadIdx.x;
    const int wave = tid >> 5;
    const int lane = tid & 31;
    const int ln   = lane & 15;
    const int g    = lane >> 4;
    const int bh   = blockIdx.y;  // b*H + h
    const int b    = bh >> 4;
    const int h    = bh & 15;
    const int q0   = blockIdx.x * 64;

    const _Float16* qb = qh + (size_t)bh * SEQ * HEADDIM;
    const _Float16* kb = kh + (size_t)bh * SEQ * HEADDIM;
    const _Float16* vb = vh + (size_t)bh * SEQ * HEADDIM;

    // Q fragments held in registers for the whole K loop.
    const int qrow = q0 + wave * 16 + ln;
    v16h aQ0 = load_frag_a(qb + (size_t)qrow * HEADDIM, g);
    v16h aQ1 = load_frag_a(qb + (size_t)qrow * HEADDIM + 32, g);

    // All-ones B fragment (layout independent) for WMMA row sums.
    HalfVec onef;
#pragma unroll
    for (int i = 0; i < 16; ++i) onef.h[i] = (_Float16)1.0f;

    v8f o[4];
#pragma unroll
    for (int t = 0; t < 4; ++t) o[t] = v8f_zero();
    float mrow[8], lrow[8];
#pragma unroll
    for (int r = 0; r < 8; ++r) { mrow[r] = -1e30f; lrow[r] = 0.f; }

    const float scale = 0.125f;  // 1/sqrt(64)

    // Staging coords: 2 threads per key row, 64 bytes each.
    const int key = tid >> 1, cb = (tid & 1) * 32;
    const unsigned vldsoff = lds_addr32(&Vs[key * 72 + cb]);
    const unsigned long long vbase = (unsigned long long)vb;

    for (int kt = 0; kt < SEQ; kt += 64) {
        // V tile (natural) via async copy; K tile transposed manually.
        async_copy64(vldsoff,
                     (unsigned)(((size_t)(kt + key) * HEADDIM + cb) * 2),
                     vbase);
        {
            union { uint4 q[4]; _Float16 h[32]; } tk;
            const _Float16* ks = kb + (size_t)(kt + key) * HEADDIM + cb;
            tk.q[0] = *reinterpret_cast<const uint4*>(ks);
            tk.q[1] = *reinterpret_cast<const uint4*>(ks + 8);
            tk.q[2] = *reinterpret_cast<const uint4*>(ks + 16);
            tk.q[3] = *reinterpret_cast<const uint4*>(ks + 24);
#pragma unroll
            for (int i = 0; i < 32; ++i) Kt[(cb + i) * 72 + key] = tk.h[i];
        }
        wait_async0();
        __syncthreads();

        // S = Q @ K^T : 4 key-chunk score tiles, 2 dim-chunk WMMAs each.
        v8f sc[4];
#pragma unroll
        for (int nch = 0; nch < 4; ++nch) {
            v8f s = v8f_zero();
            s = wmma_f16(aQ0, load_frag_b(&Kt[nch * 16 + ln], g), s);
            s = wmma_f16(aQ1, load_frag_b(&Kt[32 * 72 + nch * 16 + ln], g), s);
            sc[nch] = s * scale;
        }

        // Online softmax: running max (butterfly) + exp; sums done by WMMA.
        float alpha[8];
#pragma unroll
        for (int r = 0; r < 8; ++r) {
            float t = fmaxf(fmaxf(sc[0][r], sc[1][r]),
                            fmaxf(sc[2][r], sc[3][r]));
#pragma unroll
            for (int off = 8; off >= 1; off >>= 1)
                t = fmaxf(t, __shfl_xor(t, off, 16));
            float mnew = fmaxf(mrow[r], t);
            alpha[r]   = __expf(mrow[r] - mnew);
            mrow[r]    = mnew;
            sc[0][r] = __expf(sc[0][r] - mnew);
            sc[1][r] = __expf(sc[1][r] - mnew);
            sc[2][r] = __expf(sc[2][r] - mnew);
            sc[3][r] = __expf(sc[3][r] - mnew);
        }

        // C-layout P -> A-layout via per-wave LDS round trip.
        _Float16* pw = &Ps[wave * 16 * 72];
#pragma unroll
        for (int r = 0; r < 8; ++r) {
            int prow = r + g * 8;
#pragma unroll
            for (int nch = 0; nch < 4; ++nch)
                pw[prow * 72 + nch * 16 + ln] = (_Float16)sc[nch][r];
        }
        asm volatile("s_wait_dscnt 0x0" ::: "memory");
        v16h aP0 = load_frag_a(&pw[ln * 72], g);       // keys 0..31
        v16h aP1 = load_frag_a(&pw[ln * 72 + 32], g);  // keys 32..63

        // Row sums of the f16 P actually used for PV: lsum = P @ ones.
        v8f lsum = wmma_f16(aP0, onef.v, v8f_zero());
        lsum = wmma_f16(aP1, onef.v, lsum);
#pragma unroll
        for (int r = 0; r < 8; ++r) lrow[r] = lrow[r] * alpha[r] + lsum[r];
#pragma unroll
        for (int t = 0; t < 4; ++t)
#pragma unroll
            for (int r = 0; r < 8; ++r) o[t][r] *= alpha[r];

        // O += P @ V : 4 dim tiles x 2 key chunks.
#pragma unroll
        for (int t = 0; t < 4; ++t) {
            o[t] = wmma_f16(aP0, load_frag_b(&Vs[t * 16 + ln], g), o[t]);
            o[t] = wmma_f16(aP1, load_frag_b(&Vs[32 * 72 + t * 16 + ln], g),
                            o[t]);
        }
        __syncthreads();
    }

    // Normalize and store to attn [B, S, H*64] (feature = h*64 + dh).
    float inv[8];
#pragma unroll
    for (int r = 0; r < 8; ++r) inv[r] = 1.0f / lrow[r];
    _Float16* ab = attn +
                   ((size_t)b * SEQ + q0 + wave * 16 + g * 8) * DMODEL +
                   h * HEADDIM + ln;
#pragma unroll
    for (int t = 0; t < 4; ++t)
#pragma unroll
        for (int r = 0; r < 8; ++r)
            ab[(size_t)r * DMODEL + t * 16] = (_Float16)(o[t][r] * inv[r]);
}

// ---------------------------------------------------------------------------
// Launch
// ---------------------------------------------------------------------------
extern "C" void kernel_launch(void* const* d_in, const int* in_sizes, int n_in,
                              void* d_out, int out_size, void* d_ws,
                              size_t ws_size, hipStream_t stream) {
    const float* x     = (const float*)d_in[0];
    const float* w_qkv = (const float*)d_in[1];
    const float* b_qkv = (const float*)d_in[2];
    const float* w_out = (const float*)d_in[3];
    const float* b_out = (const float*)d_in[4];
    float* out = (float*)d_out;

    const size_t NX  = (size_t)BATCH * SEQ * DMODEL;  // 8,388,608
    const size_t NWQ = (size_t)3 * DMODEL * DMODEL;   // 3,145,728
    const size_t NWO = (size_t)DMODEL * DMODEL;       // 1,048,576

    char* ws = (char*)d_ws;
    size_t off = 0;
    auto take = [&](size_t bytes) {
        char* p = ws + off;
        off = (off + bytes + 255) & ~(size_t)255;
        return p;
    };
    _Float16* xh    = (_Float16*)take(NX * 2);
    _Float16* wqkvh = (_Float16*)take(NWQ * 2);
    _Float16* wouth = (_Float16*)take(NWO * 2);
    _Float16* qh    = (_Float16*)take(NX * 2);
    _Float16* kh    = (_Float16*)take(NX * 2);
    _Float16* vh    = (_Float16*)take(NX * 2);
    _Float16* attn  = (_Float16*)take(NX * 2);

    cvt_f32_f16_kernel<<<(int)((NX + 255) / 256), 256, 0, stream>>>(x, xh, (int)NX);
    cvt_f32_f16_kernel<<<(int)((NWQ + 255) / 256), 256, 0, stream>>>(w_qkv, wqkvh, (int)NWQ);
    cvt_f32_f16_kernel<<<(int)((NWO + 255) / 256), 256, 0, stream>>>(w_out, wouth, (int)NWO);

    dim3 gQKV(3 * DMODEL / 64, BATCH * SEQ / 128);  // (48, 64)
    wmma_gemm_kernel<1><<<gQKV, 256, 0, stream>>>(xh, wqkvh, b_qkv,
                                                  BATCH * SEQ, 3 * DMODEL,
                                                  DMODEL, nullptr, qh, kh, vh);

    size_t npairs = 2ull * BATCH * NHEADS * SEQ * 32;  // 8,388,608
    rope_kernel<<<(int)((npairs + 255) / 256), 256, 0, stream>>>(qh, kh);

    dim3 gFA(SEQ / 64, BATCH * NHEADS);  // (32, 64)
    flash_attn_kernel<<<gFA, 128, 0, stream>>>(qh, kh, vh, attn);

    dim3 gOut(DMODEL / 64, BATCH * SEQ / 128);  // (16, 64)
    wmma_gemm_kernel<0><<<gOut, 256, 0, stream>>>(attn, wouth, b_out,
                                                  BATCH * SEQ, DMODEL, DMODEL,
                                                  out, nullptr, nullptr,
                                                  nullptr);
}
